// Matcher_49503793054049
// MI455X (gfx1250) — matrix operand order repaired and checked by reading, not compiled, not run on hardware
//
#include <hip/hip_runtime.h>

#define B_   8
#define C_   512
#define H_   48
#define W_   48
#define HW_  2304

typedef __bf16        v16bf __attribute__((ext_vector_type(16)));
typedef __bf16        v4bf  __attribute__((ext_vector_type(4)));
typedef float         v8f   __attribute__((ext_vector_type(8)));
typedef float         f32x4 __attribute__((ext_vector_type(4)));
typedef unsigned int  u32x4 __attribute__((ext_vector_type(4)));

union AFrag { u32x4 q[2]; v16bf v; };

#define MSPLIT  6
#define MT_PER  24              // 144 m-tiles / 6 splits
#define NW      8               // waves per workgroup (wave32)
#define BLK     (NW * 32)       // 256 threads
#define NCOLS   (NW * 16)       // 128 output columns per workgroup
#define NBLOCKS (HW_ / NCOLS)   // 18  -> bank-tile L2 re-read factor 18x (was 36x)
#define LDS_PAD 8
#define LDS_ROW (C_ + LDS_PAD)  // 520 bf16/row -> 1040B stride, LDS banks spread

// ---------------------------------------------------------------------------
// Init fine channels (0..3) to -inf so atomic-max merging is correct.
// ---------------------------------------------------------------------------
__global__ void init_fine_kernel(float* __restrict__ out) {
    int idx = blockIdx.x * blockDim.x + threadIdx.x;   // 8*4*2304 = 73728
    if (idx >= B_ * 4 * HW_) return;
    int b  = idx / (4 * HW_);
    int r  = idx % (4 * HW_);
    int ch = r / HW_;
    int n  = r % HW_;
    out[((size_t)b * 10 + ch) * HW_ + n] = -__builtin_inff();
}

// ---------------------------------------------------------------------------
// Fine: 4 fused GEMM + (mask) + max-over-m, bf16 WMMA with f32 accumulate.
// grid = (18 n-blocks, 6 m-splits, 8 batches), block = 256 (8 waves)
// ---------------------------------------------------------------------------
__global__ __launch_bounds__(BLK) void fine_kernel(
    const float* __restrict__ key,  // [B, C, HW]
    const float* __restrict__ sds,  // [B, HW(m), HW(n)]
    const float* __restrict__ gbg, const float* __restrict__ gfg,
    const float* __restrict__ lbg, const float* __restrict__ lfg,
    float* __restrict__ out)        // [B, 10, HW]
{
    __shared__ __bf16 ldsA[4][16][LDS_ROW];   // 4 bank tiles, 66.5 KB

    const int tid  = threadIdx.x;
    const int wave = tid >> 5;
    const int lane = tid & 31;
    const int hi   = lane >> 4;     // half-wave select
    const int lo   = lane & 15;
    const int b    = blockIdx.z;
    const int n    = blockIdx.x * NCOLS + wave * 16 + lo;   // output column

    const float* banks[4] = {
        gbg + (size_t)b * HW_ * C_,
        gfg + (size_t)b * HW_ * C_,
        lbg + (size_t)b * HW_ * C_,
        lfg + (size_t)b * HW_ * C_ };

    // ---- resident B fragments: 16 key columns, all of C, bf16 (128 VGPRs) ----
    // B layout (32x16, K-major): lane L: N = L&15, element e <-> K = hi*16 + e
    const float* kb = key + (size_t)b * C_ * HW_;
    v16bf bk[16];
#pragma unroll
    for (int kc = 0; kc < 16; ++kc) {
        v16bf t;
#pragma unroll
        for (int e = 0; e < 16; ++e) {
            const int K = kc * 32 + hi * 16 + e;
            t[e] = (__bf16)kb[(size_t)K * HW_ + n];
        }
        bk[kc] = t;
    }

    const float NEG = -__builtin_inff();
    float mx4[4] = { NEG, NEG, NEG, NEG };    // scalar running max per bank

    for (int it = 0; it < MT_PER; ++it) {
        const int mt = blockIdx.y * MT_PER + it;
        const int m0 = mt * 16;

        __syncthreads();
        // ---- cooperative fp32->bf16 conversion of 4 bank tiles into LDS ----
        // 8 loads in flight per thread before converting (latency batching)
#pragma unroll
        for (int q = 0; q < 4; ++q) {
            const float* bp = banks[q] + (size_t)m0 * C_;
            f32x4 t[8];
#pragma unroll
            for (int j = 0; j < 8; ++j) {
                const int elem = (j * BLK + tid) * 4;     // 0..8191
                t[j] = *reinterpret_cast<const f32x4*>(bp + elem);
            }
#pragma unroll
            for (int j = 0; j < 8; ++j) {
                const int elem = (j * BLK + tid) * 4;
                const int row  = elem >> 9;               // / 512
                const int col  = elem & (C_ - 1);
                *reinterpret_cast<v4bf*>(&ldsA[q][row][col]) =
                    __builtin_convertvector(t[j], v4bf);
            }
        }
        __syncthreads();

        v8f acc[4];
#pragma unroll
        for (int q = 0; q < 4; ++q)
#pragma unroll
            for (int r = 0; r < 8; ++r) acc[q][r] = 0.0f;

        // A layout (16x32): lane L: M = L&15, runs K = hi*16+{0..15}
        const int row = lo;
#pragma unroll
        for (int kc = 0; kc < 16; ++kc) {
            const int base = kc * 32 + hi * 16;
            AFrag f[4];
#pragma unroll
            for (int q = 0; q < 4; ++q) {                 // 8 ds_loads first
                f[q].q[0] = *reinterpret_cast<const u32x4*>(&ldsA[q][row][base]);
                f[q].q[1] = *reinterpret_cast<const u32x4*>(&ldsA[q][row][base + 8]);
            }
#pragma unroll
            for (int q = 0; q < 4; ++q) {                 // then 4 independent WMMAs
                acc[q] = __builtin_amdgcn_wmma_f32_16x16x32_bf16(
                    false, f[q].v, false, bk[kc], (short)0, acc[q], false, false);
            }
        }

        // ---- epilogue: mask local banks, then per-tile max into scalars ----
        // D layout: element r, lane L -> m = m0 + r + 8*hi, col n
        const float* mcol = sds + (size_t)b * HW_ * HW_ + (size_t)m0 * HW_ + n;
        float w[8];
#pragma unroll
        for (int r = 0; r < 8; ++r)                       // 8 loads in flight
            w[r] = mcol[(size_t)(r + 8 * hi) * HW_];
#pragma unroll
        for (int r = 0; r < 8; ++r) {
            acc[2][r] *= w[r];
            acc[3][r] *= w[r];
        }
#pragma unroll
        for (int q = 0; q < 4; ++q) {
            float v = acc[q][0];
#pragma unroll
            for (int r = 1; r < 8; ++r) v = fmaxf(v, acc[q][r]);
            mx4[q] = fmaxf(mx4[q], v);
        }
    }

    // ---- half-wave swap, merge partial max into out ----
#pragma unroll
    for (int q = 0; q < 4; ++q) {
        float v = mx4[q];
        v = fmaxf(v, __shfl_xor(v, 16, 32));
        if (hi == 0) {
            float* dst = out + ((size_t)b * 10 + q) * HW_ + n;
            __hip_atomic_fetch_max(dst, v, __ATOMIC_RELAXED, __HIP_MEMORY_SCOPE_AGENT);
        }
    }
}

// ---------------------------------------------------------------------------
// Coarse: 6 channels of bank[1,C] . key[C, HW]  (trivial, memory bound)
// grid = (HW/256, B), block = 256
// ---------------------------------------------------------------------------
__global__ void coarse_kernel(
    const float* __restrict__ key,
    const float* __restrict__ obg, const float* __restrict__ ofg,
    const float* __restrict__ sbg, const float* __restrict__ sfg,
    const float* __restrict__ lobg, const float* __restrict__ lofg,
    float* __restrict__ out)
{
    const int b = blockIdx.y;
    const int n = blockIdx.x * 256 + threadIdx.x;
    const float* kb = key + (size_t)b * C_ * HW_;
    float a0 = 0.f, a1 = 0.f, a2 = 0.f, a3 = 0.f, a4 = 0.f, a5 = 0.f;
    for (int c = 0; c < C_; ++c) {
        const float kv = kb[(size_t)c * HW_ + n];
        a0 += obg[b * C_ + c] * kv;
        a1 += ofg[b * C_ + c] * kv;
        a2 += sbg[b * C_ + c] * kv;
        a3 += sfg[b * C_ + c] * kv;
        a4 += lobg[b * C_ + c] * kv;
        a5 += lofg[b * C_ + c] * kv;
    }
    float* o = out + (size_t)b * 10 * HW_ + n;
    o[4 * HW_] = a0;  o[5 * HW_] = a1;
    o[6 * HW_] = a2;  o[7 * HW_] = a3;
    o[8 * HW_] = a4;  o[9 * HW_] = a5;
}

// ---------------------------------------------------------------------------
extern "C" void kernel_launch(void* const* d_in, const int* in_sizes, int n_in,
                              void* d_out, int out_size, void* d_ws, size_t ws_size,
                              hipStream_t stream) {
    (void)in_sizes; (void)n_in; (void)out_size; (void)d_ws; (void)ws_size;
    const float* key  = (const float*)d_in[0];
    const float* sds  = (const float*)d_in[1];
    const float* gbg  = (const float*)d_in[2];
    const float* gfg  = (const float*)d_in[3];
    const float* lbg  = (const float*)d_in[4];
    const float* lfg  = (const float*)d_in[5];
    const float* obg  = (const float*)d_in[6];
    const float* ofg  = (const float*)d_in[7];
    const float* sbg  = (const float*)d_in[8];
    const float* sfg  = (const float*)d_in[9];
    const float* lobg = (const float*)d_in[10];
    const float* lofg = (const float*)d_in[11];
    float* out = (float*)d_out;

    hipLaunchKernelGGL(init_fine_kernel, dim3((B_ * 4 * HW_ + 255) / 256), dim3(256),
                       0, stream, out);
    hipLaunchKernelGGL(fine_kernel, dim3(NBLOCKS, MSPLIT, B_), dim3(BLK),
                       0, stream, key, sds, gbg, gfg, lbg, lfg, out);
    hipLaunchKernelGGL(coarse_kernel, dim3(HW_ / 256, B_), dim3(256),
                       0, stream, key, obg, ofg, sbg, sfg, lobg, lofg, out);
}
